// SelectiveSSM_47390669144407
// MI455X (gfx1250) — matrix operand order repaired and checked by reading, not compiled, not run on hardware
//
#include <hip/hip_runtime.h>

// ---------------- problem constants (match reference) ----------------
#define D_MODEL   1024
#define D_STATE   16
#define D_CONV    4
#define D_INNER   1024          // EXPAND = 1
#define DT_RANK   64
#define BATCH     2
#define SEQ       2048
#define MTOK      (BATCH * SEQ)             // 4096 tokens
#define N_XDBL    (DT_RANK + 2 * D_STATE)   // 96

// ---------------- WMMA vector types (gfx1250, wave32) ----------------
typedef __bf16 v16bf __attribute__((ext_vector_type(16)));
typedef __bf16 v8bf  __attribute__((ext_vector_type(8)));
typedef float  v8f   __attribute__((ext_vector_type(8)));

union Frag16 { v16bf v; v8bf h[2]; };
union Acc8   { v8f v; float f[8]; };

// ---------------- helpers ----------------
__device__ __forceinline__ float silu_f(float x) {
    return x / (1.0f + __expf(-x));
}
__device__ __forceinline__ float softplus_f(float x) {
    return (x > 20.0f) ? x : log1pf(__expf(x));
}

// gfx1250 async copy global -> LDS (ASYNCcnt-tracked, ISA ch.10/15.18)
__device__ __forceinline__ void async_load_b128(unsigned lds_addr, unsigned voff,
                                                const void* sbase) {
    asm volatile("global_load_async_to_lds_b128 %0, %1, %2"
                 :: "v"(lds_addr), "v"(voff),
                    "s"((unsigned long long)sbase)
                 : "memory");
}
__device__ __forceinline__ void wait_asynccnt0() {
    asm volatile("s_wait_asynccnt 0x0" ::: "memory");
}
// generic LDS pointer -> 32-bit LDS byte offset (low 32 bits per aperture map)
__device__ __forceinline__ unsigned lds_off(const void* p) {
    return (unsigned)(unsigned long long)p;
}

// ---------------- generic f32 -> bf16 cast ----------------
__global__ void k_cast_bf16(const float* __restrict__ src,
                            __bf16* __restrict__ dst, int n) {
    int i = blockIdx.x * blockDim.x + threadIdx.x;
    if (i < n) dst[i] = (__bf16)src[i];
}

// extract dt_raw (cols 0..63 of x_dbl[M,96]) as bf16 [M,64]
__global__ void k_extract_dtraw(const float* __restrict__ xdbl,
                                __bf16* __restrict__ dtb, int n) {
    int i = blockIdx.x * blockDim.x + threadIdx.x;
    if (i >= n) return;
    int m = i >> 6, r = i & 63;
    dtb[i] = (__bf16)xdbl[m * N_XDBL + r];
}

// ================= tiled WMMA GEMM with async-LDS A staging =================
//   C[M,N] = A[M,K] * B[N,K]^T        (A,B bf16 row-major; C f32 row-major)
// Block: 256 thr = 8 waves as 2(M) x 4(N); block tile 32 x 256.
// Wave tile: 16 x 64 (4 accumulators, one A frag reused 4x).
// A panel (32 rows x 32 halfs) double-buffered in LDS via
// global_load_async_to_lds_b128 + s_wait_asynccnt (wave 0 is the producer).
// mode 0: plain store; mode 1: C = softplus(acc + bias[col]).
__global__ void k_gemm_wmma_tiled(const __bf16* __restrict__ A,
                                  const __bf16* __restrict__ B,
                                  float* __restrict__ C,
                                  int M, int N, int K,
                                  int lda, int ldb, int ldc,
                                  const float* __restrict__ bias, int mode) {
    __shared__ __bf16 sA[2][32][32];      // 2 x 2KB double buffer

    const int lane = threadIdx.x & 31;
    const int wave = threadIdx.x >> 5;
    const int wm   = wave >> 2;           // 0..1   (M sub-tile)
    const int wn   = wave & 3;            // 0..3   (N sub-tile)
    const int half = lane >> 4;           // 0: lanes 0-15, 1: lanes 16-31
    const int lrow = lane & 15;

    const int mBlk = blockIdx.y * 32;
    const int m0   = mBlk + wm * 16;
    const int n0   = blockIdx.x * 256 + wn * 64;

    const unsigned sAbase = lds_off(&sA[0][0][0]);

    // producer: kick one 32x32-half A chunk (128 x b128) into LDS buffer `buf`
    auto kickA = [&](int k0, int buf) {
#pragma unroll
        for (int t = 0; t < 4; ++t) {
            int c   = t * 32 + lane;              // 0..127 16-byte chunks
            int row = c >> 2;
            int kq  = (c & 3) * 8;
            unsigned voff = (unsigned)(((mBlk + row) * lda + k0 + kq) * 2);
            async_load_b128(sAbase + (unsigned)(buf * 2048 + c * 16), voff, A);
        }
    };

    v8f acc[4] = {};
    if (wave == 0) kickA(0, 0);

    for (int k0 = 0, it = 0; k0 < K; k0 += 32, ++it) {
        const int buf = it & 1;
        if (wave == 0) wait_asynccnt0();          // chunk `it` landed in LDS
        __syncthreads();
        if (wave == 0 && k0 + 32 < K) kickA(k0 + 32, buf ^ 1);

        Frag16 a;                                  // ds_read_b128 x2
        a.h[0] = *(const v8bf*)&sA[buf][wm * 16 + lrow][half * 8];
        a.h[1] = *(const v8bf*)&sA[buf][wm * 16 + lrow][16 + half * 8];

#pragma unroll
        for (int nt = 0; nt < 4; ++nt) {
            const __bf16* Bp = B + (size_t)(n0 + nt * 16 + lrow) * ldb
                                 + half * 8 + k0;
            Frag16 bfr;
            bfr.h[0] = *(const v8bf*)(Bp);
            bfr.h[1] = *(const v8bf*)(Bp + 16);
            if (k0 + 32 < K) __builtin_prefetch(Bp + 32, 0, 3);
            acc[nt] = __builtin_amdgcn_wmma_f32_16x16x32_bf16(
                false, a.v, false, bfr.v, (short)0, acc[nt], false, false);
        }
    }

    // C tile layout: lane L, reg r -> row m0 + r + (L/16)*8, col base + L%16
#pragma unroll
    for (int nt = 0; nt < 4; ++nt) {
        Acc8 u; u.v = acc[nt];
        const int col = n0 + nt * 16 + lrow;
        const float bv = (mode == 1) ? bias[col] : 0.0f;
#pragma unroll
        for (int r = 0; r < 8; ++r) {
            int row = m0 + r + half * 8;
            float val = u.f[r];
            if (mode == 1) val = softplus_f(val + bv);
            C[(size_t)row * ldc + col] = val;
        }
    }
}

// ============== simple WMMA GEMM (one 16x16 tile per wave) ==============
// used for the skinny x_dbl projection (N = 96)
__global__ void k_gemm_wmma_simple(const __bf16* __restrict__ A,
                                   const __bf16* __restrict__ B,
                                   float* __restrict__ C,
                                   int M, int N, int K,
                                   int lda, int ldb, int ldc) {
    const int lane = threadIdx.x & 31;
    const int wave = threadIdx.x >> 5;
    const int wavesPerBlk = blockDim.x >> 5;
    const int n0 = (blockIdx.x * wavesPerBlk + wave) * 16;
    const int m0 = blockIdx.y * 16;
    if (n0 >= N || m0 >= M) return;

    const int half = lane >> 4;
    const int lrow = lane & 15;
    const __bf16* Aptr = A + (size_t)(m0 + lrow) * lda + half * 8;
    const __bf16* Bptr = B + (size_t)(n0 + lrow) * ldb + half * 8;

    v8f acc = {};
    for (int k0 = 0; k0 < K; k0 += 32) {
        Frag16 a, b;
        a.h[0] = *(const v8bf*)(Aptr + k0);
        a.h[1] = *(const v8bf*)(Aptr + k0 + 16);
        b.h[0] = *(const v8bf*)(Bptr + k0);
        b.h[1] = *(const v8bf*)(Bptr + k0 + 16);
        if (k0 + 32 < K) {
            __builtin_prefetch(Aptr + k0 + 32, 0, 3);
            __builtin_prefetch(Bptr + k0 + 32, 0, 3);
        }
        acc = __builtin_amdgcn_wmma_f32_16x16x32_bf16(
            false, a.v, false, b.v, (short)0, acc, false, false);
    }
    Acc8 u; u.v = acc;
    const int col = n0 + lrow;
#pragma unroll
    for (int r = 0; r < 8; ++r)
        C[(size_t)(m0 + r + half * 8) * ldc + col] = u.f[r];
}

// ---------------- causal depthwise conv (k=4) + bias + SiLU ----------------
__global__ void k_conv_silu(const float* __restrict__ xz,
                            const float* __restrict__ convw,
                            const float* __restrict__ convb,
                            float* __restrict__ xcf,
                            __bf16* __restrict__ xcb) {
    int idx = blockIdx.x * blockDim.x + threadIdx.x;
    if (idx >= MTOK * D_INNER) return;
    int d  = idx & (D_INNER - 1);
    int ml = idx / D_INNER;
    int l  = ml & (SEQ - 1);
    int b  = ml / SEQ;
    float acc = convb[d];
#pragma unroll
    for (int j = 0; j < D_CONV; ++j) {
        int ls = l - (D_CONV - 1) + j;
        if (ls >= 0)
            acc += convw[d * D_CONV + j] *
                   xz[(size_t)(b * SEQ + ls) * (2 * D_INNER) + d];
    }
    float s = silu_f(acc);
    xcf[idx] = s;
    xcb[idx] = (__bf16)s;
}

// ---------------- selective scan ----------------
// 16 lanes per channel (lane%16 = state n). 16 channels / 256-thread block.
__global__ void k_scan(const float* __restrict__ dt,
                       const float* __restrict__ xc,
                       const float* __restrict__ xdbl,
                       const float* __restrict__ xz,
                       const float* __restrict__ Alog,
                       const float* __restrict__ Dvec,
                       __bf16* __restrict__ ybf) {
    const int n = threadIdx.x & 15;
    const int d = blockIdx.x * 16 + (threadIdx.x >> 4);
    const int b = blockIdx.y;
    const float A  = -__expf(Alog[d * D_STATE + n]);
    const float Dv = Dvec[d];
    float h = 0.0f;
    const size_t tok0 = (size_t)b * SEQ;
    for (int l = 0; l < SEQ; ++l) {
        const size_t m = tok0 + l;
        const float dtv = dt[m * D_INNER + d];
        const float xv  = xc[m * D_INNER + d];
        const float Bv  = xdbl[m * N_XDBL + DT_RANK + n];
        const float Cv  = xdbl[m * N_XDBL + DT_RANK + D_STATE + n];
        h = __expf(dtv * A) * h + (dtv * xv) * Bv;
        float p = h * Cv;
        p += __shfl_xor(p, 1, 16);
        p += __shfl_xor(p, 2, 16);
        p += __shfl_xor(p, 4, 16);
        p += __shfl_xor(p, 8, 16);
        if (n == 0) {
            const float zv = xz[m * (2 * D_INNER) + D_INNER + d];
            const float y  = (p + xv * Dv) * silu_f(zv);
            ybf[m * D_INNER + d] = (__bf16)y;
        }
    }
}

// ---------------- host launch ----------------
extern "C" void kernel_launch(void* const* d_in, const int* in_sizes, int n_in,
                              void* d_out, int out_size, void* d_ws, size_t ws_size,
                              hipStream_t stream) {
    const float* x      = (const float*)d_in[0];   // [2,2048,1024]
    const float* W_in   = (const float*)d_in[1];   // [2048,1024]
    const float* conv_w = (const float*)d_in[2];   // [1024,1,4]
    const float* conv_b = (const float*)d_in[3];   // [1024]
    const float* W_x    = (const float*)d_in[4];   // [96,1024]
    const float* W_dt   = (const float*)d_in[5];   // [1024,64]
    const float* b_dt   = (const float*)d_in[6];   // [1024]
    const float* A_log  = (const float*)d_in[7];   // [1024,16]
    const float* Dvec   = (const float*)d_in[8];   // [1024]
    const float* W_out  = (const float*)d_in[9];   // [1024,1024]
    float* out = (float*)d_out;                    // [2,2048,1024] fp32

    char* ws = (char*)d_ws;
    size_t off = 0;
    auto alloc = [&](size_t bytes) -> void* {
        void* p = ws + off;
        off = (off + bytes + 255) & ~(size_t)255;
        return p;
    };
    __bf16* Xbf    = (__bf16*)alloc((size_t)MTOK * D_MODEL * 2);
    __bf16* Winbf  = (__bf16*)alloc((size_t)2 * D_INNER * D_MODEL * 2);
    __bf16* Wxbf   = (__bf16*)alloc((size_t)N_XDBL * D_INNER * 2);
    __bf16* Wdtbf  = (__bf16*)alloc((size_t)D_INNER * DT_RANK * 2);
    __bf16* Woutbf = (__bf16*)alloc((size_t)D_MODEL * D_INNER * 2);
    float*  xzf    = (float*) alloc((size_t)MTOK * 2 * D_INNER * 4);
    float*  xcf    = (float*) alloc((size_t)MTOK * D_INNER * 4);
    __bf16* xcb    = (__bf16*)alloc((size_t)MTOK * D_INNER * 2);
    float*  xdbl   = (float*) alloc((size_t)MTOK * N_XDBL * 4);
    __bf16* dtrb   = (__bf16*)alloc((size_t)MTOK * DT_RANK * 2);
    float*  dtf    = (float*) alloc((size_t)MTOK * D_INNER * 4);
    __bf16* ybf    = (__bf16*)alloc((size_t)MTOK * D_INNER * 2);
    (void)ws_size; (void)n_in; (void)in_sizes; (void)out_size;

    auto cast = [&](const float* s, __bf16* d, int n) {
        k_cast_bf16<<<(n + 255) / 256, 256, 0, stream>>>(s, d, n);
    };
    cast(x,     Xbf,    MTOK * D_MODEL);
    cast(W_in,  Winbf,  2 * D_INNER * D_MODEL);
    cast(W_x,   Wxbf,   N_XDBL * D_INNER);
    cast(W_dt,  Wdtbf,  D_INNER * DT_RANK);
    cast(W_out, Woutbf, D_MODEL * D_INNER);

    // 1) in_proj: xz[4096,2048] = Xbf @ W_in^T   (tiled, async-LDS A panel)
    k_gemm_wmma_tiled<<<dim3((2 * D_INNER) / 256, MTOK / 32), 256, 0, stream>>>(
        Xbf, Winbf, xzf, MTOK, 2 * D_INNER, D_MODEL,
        D_MODEL, D_MODEL, 2 * D_INNER, nullptr, 0);

    // 2) depthwise causal conv + SiLU
    k_conv_silu<<<(MTOK * D_INNER + 255) / 256, 256, 0, stream>>>(
        xzf, conv_w, conv_b, xcf, xcb);

    // 3) x_dbl[4096,96] = xc @ W_x^T   (simple kernel, 6 waves cover N=96)
    k_gemm_wmma_simple<<<dim3(1, MTOK / 16), 192, 0, stream>>>(
        xcb, Wxbf, xdbl, MTOK, N_XDBL, D_INNER,
        D_INNER, D_INNER, N_XDBL);

    // 4) dt = softplus(dt_raw @ W_dt^T + b_dt)   (tiled, K=64)
    k_extract_dtraw<<<(MTOK * DT_RANK + 255) / 256, 256, 0, stream>>>(
        xdbl, dtrb, MTOK * DT_RANK);
    k_gemm_wmma_tiled<<<dim3(D_INNER / 256, MTOK / 32), 256, 0, stream>>>(
        dtrb, Wdtbf, dtf, MTOK, D_INNER, DT_RANK,
        DT_RANK, DT_RANK, D_INNER, b_dt, 1);

    // 5) selective scan + D skip + SiLU(z) gate -> y (bf16 for out_proj)
    k_scan<<<dim3(D_INNER / 16, BATCH), 256, 0, stream>>>(
        dtf, xcf, xdbl, xzf, A_log, Dvec, ybf);

    // 6) out = y @ W_out^T  -> d_out (fp32)
    k_gemm_wmma_tiled<<<dim3(D_MODEL / 256, MTOK / 32), 256, 0, stream>>>(
        ybf, Woutbf, out, MTOK, D_MODEL, D_INNER,
        D_INNER, D_INNER, D_MODEL, nullptr, 0);
}
